// LocalAttention_39865886441710
// MI455X (gfx1250) — matrix operand order repaired and checked by reading, not compiled, not run on hardware
//
#include <hip/hip_runtime.h>

typedef __bf16 bf16_t;
typedef __attribute__((ext_vector_type(8)))  __bf16 bf16x8;
typedef __attribute__((ext_vector_type(16))) __bf16 bf16x16;
typedef __attribute__((ext_vector_type(8)))  float  f32x8;

static __device__ __forceinline__ bf16x8 ld8(const bf16_t* p) {
  return *(const bf16x8*)p;
}
static __device__ __forceinline__ bf16x16 join16(bf16x8 lo, bf16x8 hi) {
  bf16x16 r;
#pragma unroll
  for (int i = 0; i < 8; ++i) { r[i] = lo[i]; r[i + 8] = hi[i]; }
  return r;
}
static __device__ __forceinline__ f32x8 wmma_bf16(bf16x16 a, bf16x16 b, f32x8 c) {
  // D = A(16x32) * B(32x16) + C, f32 accumulate
  return __builtin_amdgcn_wmma_f32_16x16x32_bf16(false, a, false, b, (short)0, c,
                                                 false, false);
}
static __device__ __forceinline__ f32x8 zero8() {
  f32x8 r;
#pragma unroll
  for (int i = 0; i < 8; ++i) r[i] = 0.0f;
  return r;
}

// CDNA5 async global->LDS copy: stages 32 contiguous bytes per lane.
// IOFFSET is added to BOTH the LDS and global addresses (ISA 10, async ops),
// so one base pair covers both 16B chunks.
static __device__ __forceinline__ void async_pair(unsigned lds, const bf16_t* g) {
  asm volatile("global_load_async_to_lds_b128 %0, %1, off"
               :: "v"(lds), "v"(g) : "memory");
  asm volatile("global_load_async_to_lds_b128 %0, %1, off offset:16"
               :: "v"(lds), "v"(g) : "memory");
}
static __device__ __forceinline__ void wait_async0() {
  asm volatile("s_wait_asynccnt 0x0" ::: "memory");
}
// Low 32 bits of a generic pointer into LDS = byte offset in wave's LDS alloc.
static __device__ __forceinline__ unsigned lds_addr(const void* p) {
  return (unsigned)(uintptr_t)p;
}

// ---------------------------------------------------------------------------
// Elementwise f32 -> bf16 conversion (for x)
// ---------------------------------------------------------------------------
__global__ void cvt_f32_bf16(const float* __restrict__ s, bf16_t* __restrict__ d, int n) {
  int i = blockIdx.x * 256 + threadIdx.x;
  if (i < n) d[i] = (bf16_t)s[i];
}

// ---------------------------------------------------------------------------
// Weight convert + transpose: W f32 (K=1024, N=1024) -> Wt bf16 (N, K).
// LDS-tiled 32x32 so both global read and write are coalesced.
// ---------------------------------------------------------------------------
__global__ __launch_bounds__(256) void cvt_w_t(const float* __restrict__ s,
                                               bf16_t* __restrict__ d) {
  __shared__ float t[32][33];
  const int bx = blockIdx.x & 31;   // n tile
  const int by = blockIdx.x >> 5;   // k tile
  const int k0 = by * 32, n0 = bx * 32;
  const int r  = threadIdx.x >> 3;        // 0..31
  const int c4 = (threadIdx.x & 7) * 4;   // 0..28
#pragma unroll
  for (int j = 0; j < 4; ++j)
    t[r][c4 + j] = s[(size_t)(k0 + r) * 1024 + n0 + c4 + j];
  __syncthreads();
#pragma unroll
  for (int j = 0; j < 4; ++j)
    d[(size_t)(n0 + r) * 1024 + k0 + c4 + j] = (bf16_t)t[c4 + j][r];
}

// ---------------------------------------------------------------------------
// Tiled bf16 GEMM: C(f32, 2048x1024) = A(bf16, Mx1024) * Bt(bf16, 1024x1024)^T
// Bt is stored transposed (N,K) so both LDS tiles stage with contiguous
// async b128 copies. Block tile 128x128, 8 waves (4x2), wave tile 32x64,
// K-step 32, double-buffered LDS fed by global_load_async_to_lds_b128.
// ---------------------------------------------------------------------------
constexpr int GN = 1024;
constexpr int GK = 1024;
#define LDSSTRIDE 48  // elements; 96B rows keep 16B-aligned sub-loads

__global__ __launch_bounds__(256) void gemm_bf16(const bf16_t* __restrict__ A,
                                                 const bf16_t* __restrict__ Bt,
                                                 float* __restrict__ C) {
  __shared__ bf16_t As[2][128 * LDSSTRIDE];
  __shared__ bf16_t Bs[2][128 * LDSSTRIDE];  // [n][k] layout

  const int tid  = threadIdx.x;
  const int wave = tid >> 5;
  const int lane = tid & 31;
  const int l15  = lane & 15;
  const int hi   = lane >> 4;  // 0 or 1

  const int bm = blockIdx.y * 128;
  const int bn = blockIdx.x * 128;
  const int wr = (wave >> 1) * 32;
  const int wc = (wave & 1) * 64;

  // staging geometry: each thread owns one row-half (32 bytes) of each tile
  const int srow = tid >> 1;
  const int scol = (tid & 1) * 16;
  const bf16_t* ag = A  + (size_t)(bm + srow) * GK + scol;
  const bf16_t* bg = Bt + (size_t)(bn + srow) * GK + scol;
  unsigned al0 = lds_addr(&As[0][srow * LDSSTRIDE + scol]);
  unsigned al1 = lds_addr(&As[1][srow * LDSSTRIDE + scol]);
  unsigned bl0 = lds_addr(&Bs[0][srow * LDSSTRIDE + scol]);
  unsigned bl1 = lds_addr(&Bs[1][srow * LDSSTRIDE + scol]);

  f32x8 acc[2][4];
#pragma unroll
  for (int i = 0; i < 2; ++i)
#pragma unroll
    for (int j = 0; j < 4; ++j) acc[i][j] = zero8();

  // prologue: stage k0=0 into buffer 0
  async_pair(al0, ag);
  async_pair(bl0, bg);
  wait_async0();
  __syncthreads();

#pragma unroll 2
  for (int k0 = 0; k0 < GK; k0 += 32) {
    const int buf = (k0 >> 5) & 1;
    if (k0 + 32 < GK) {  // prefetch next tile into other buffer
      async_pair(buf ? al0 : al1, ag + k0 + 32);
      async_pair(buf ? bl0 : bl1, bg + k0 + 32);
    }

    bf16x16 af[2];
#pragma unroll
    for (int t = 0; t < 2; ++t) {
      int m  = wr + t * 16 + l15;
      int kb = hi * 8;
      const bf16_t* p = &As[buf][m * LDSSTRIDE + kb];
      af[t] = join16(ld8(p), ld8(p + 16));
    }
    bf16x16 bfm[4];
#pragma unroll
    for (int t = 0; t < 4; ++t) {
      int n  = wc + t * 16 + l15;
      int kb = hi * 8;
      const bf16_t* p = &Bs[buf][n * LDSSTRIDE + kb];
      bfm[t] = join16(ld8(p), ld8(p + 16));
    }
#pragma unroll
    for (int i = 0; i < 2; ++i)
#pragma unroll
      for (int j = 0; j < 4; ++j) acc[i][j] = wmma_bf16(af[i], bfm[j], acc[i][j]);

    wait_async0();     // own prefetch landed
    __syncthreads();   // everyone's landed; everyone done reading old buffer
  }

  // C stores: constant strides fold into instruction offsets
#pragma unroll
  for (int i = 0; i < 2; ++i)
#pragma unroll
    for (int j = 0; j < 4; ++j) {
      float* cp = C + (size_t)(bm + wr + i * 16 + hi * 8) * GN + bn + wc + j * 16 + l15;
#pragma unroll
      for (int e = 0; e < 8; ++e) cp[e * GN] = acc[i][j][e];
    }
}

// ---------------------------------------------------------------------------
// RoPE + relayout: src f32 (B,L,1024) -> dst bf16 (B,H,L,64)
// ---------------------------------------------------------------------------
__global__ void rope_pack(const float* __restrict__ src, const float* __restrict__ freqs,
                          bf16_t* __restrict__ dst) {
  int i  = blockIdx.x * 256 + threadIdx.x;  // 1,048,576 total
  int d2 = i & 31;
  int l  = (i >> 5) & 1023;
  int h  = (i >> 15) & 15;
  int b  = i >> 19;
  if (b >= 2) return;
  float f  = freqs[l * 32 + d2];
  float c0 = cosf(f);
  float s0 = sinf(f);
  size_t si = ((size_t)(b * 1024 + l)) * 1024 + h * 64 + 2 * d2;
  float re = src[si];
  float im = src[si + 1];
  size_t di = (((size_t)(b * 16 + h) * 1024) + l) * 64 + 2 * d2;
  dst[di]     = (bf16_t)(re * c0 - im * s0);
  dst[di + 1] = (bf16_t)(re * s0 + im * c0);
}

// ---------------------------------------------------------------------------
// V relayout: src f32 (B,L,1024) -> dst bf16 (B,H,64,LP) transposed per head,
// zero-padded: column lp maps to sequence pos l = lp - 64
// ---------------------------------------------------------------------------
#define LP 1120
__global__ void v_pack(const float* __restrict__ src, bf16_t* __restrict__ dst) {
  int i  = blockIdx.x * 256 + threadIdx.x;  // 2,293,760 total
  int lp = i % LP;
  int d  = (i / LP) & 63;
  int h  = (i / (LP * 64)) & 15;
  int b  = i / (LP * 64 * 16);
  if (b >= 2) return;
  int l = lp - 64;
  float v = (l >= 0 && l < 1024) ? src[((size_t)(b * 1024 + l)) * 1024 + h * 64 + d] : 0.0f;
  dst[(((size_t)(b * 16 + h) * 64) + d) * LP + lp] = (bf16_t)v;
}

// ---------------------------------------------------------------------------
// Fused sliding-window attention. One wave = one 16-query tile of one head.
// Window aligned to ws2 = q0-64; 96 candidate keys; causal band c in [m+1,m+64].
// scores: 6x2 WMMA; shfl softmax; P via per-wave LDS; O: 4x3 WMMA.
// ---------------------------------------------------------------------------
__global__ __launch_bounds__(256) void local_attn(const bf16_t* __restrict__ Q,
                                                  const bf16_t* __restrict__ Km,
                                                  const bf16_t* __restrict__ Vt,
                                                  bf16_t* __restrict__ AO) {
  __shared__ bf16_t PL[8][16 * 96];

  const int wave = threadIdx.x >> 5;
  const int lane = threadIdx.x & 31;
  const int l15  = lane & 15;
  const int hi   = lane >> 4;
  const int hi8  = hi * 8;

  const int tile = blockIdx.x * 8 + wave;  // 0..2047
  const int qt   = tile & 63;
  const int head = tile >> 6;  // b*16+h
  const int q0   = qt * 16;
  const int ws2  = q0 - 64;

  const bf16_t* q  = Q  + (size_t)head * 1024 * 64;
  const bf16_t* k  = Km + (size_t)head * 1024 * 64;
  const bf16_t* vt = Vt + (size_t)head * 64 * LP + 64;

  bf16x16 qf[2];
#pragma unroll
  for (int ks = 0; ks < 2; ++ks) {
    int kb = ks * 32 + hi8;
    const bf16_t* p = q + (size_t)(q0 + l15) * 64 + kb;
    qf[ks] = join16(ld8(p), ld8(p + 16));
  }

  f32x8 st[6];
#pragma unroll
  for (int t = 0; t < 6; ++t) {
    int key  = ws2 + t * 16 + l15;
    int keyc = key < 0 ? 0 : (key > 1023 ? 1023 : key);
    f32x8 a = zero8();
#pragma unroll
    for (int ks = 0; ks < 2; ++ks) {
      int kb = ks * 32 + hi * 8;
      const bf16_t* p = k + (size_t)keyc * 64 + kb;
      bf16x16 bf = join16(ld8(p), ld8(p + 16));
      a = wmma_bf16(qf[ks], bf, a);
    }
    st[t] = a;
  }

  const float scale = 0.125f;  // 1/sqrt(64)
#pragma unroll
  for (int e = 0; e < 8; ++e) {
    int m = e + hi8;
    float mx = -1e30f;
#pragma unroll
    for (int t = 0; t < 6; ++t) {
      int c = t * 16 + l15;
      bool valid = (c >= m + 1) && (c <= m + 64) && (ws2 + c >= 0);
      float s = valid ? st[t][e] * scale : -1e30f;
      st[t][e] = s;
      mx = fmaxf(mx, s);
    }
#pragma unroll
    for (int off = 1; off < 16; off <<= 1) mx = fmaxf(mx, __shfl_xor(mx, off));
    float sum = 0.0f;
#pragma unroll
    for (int t = 0; t < 6; ++t) {
      float pv = __expf(st[t][e] - mx);
      st[t][e] = pv;
      sum += pv;
    }
#pragma unroll
    for (int off = 1; off < 16; off <<= 1) sum += __shfl_xor(sum, off);
    float inv = __frcp_rn(sum);
#pragma unroll
    for (int t = 0; t < 6; ++t)
      PL[wave][m * 96 + t * 16 + l15] = (bf16_t)(st[t][e] * inv);
  }

  bf16x16 pf[3];
#pragma unroll
  for (int ks = 0; ks < 3; ++ks) {
    int kb = ks * 32 + hi8;
    const bf16_t* p = &PL[wave][l15 * 96 + kb];
    pf[ks] = join16(ld8(p), ld8(p + 16));
  }

  f32x8 oc[4];
#pragma unroll
  for (int tc = 0; tc < 4; ++tc) {
    f32x8 a = zero8();
#pragma unroll
    for (int ks = 0; ks < 3; ++ks) {
      int kb = ks * 32 + hi * 8;
      const bf16_t* p = vt + (size_t)(tc * 16 + l15) * LP + ws2 + kb;
      bf16x16 bf = join16(ld8(p), ld8(p + 16));
      a = wmma_bf16(pf[ks], bf, a);
    }
    oc[tc] = a;
  }

  const int b    = head >> 4;
  const int hcol = (head & 15) * 64;
  bf16_t* aop = AO + ((size_t)(b * 1024 + q0 + hi8)) * 1024 + hcol + l15;
#pragma unroll
  for (int tc = 0; tc < 4; ++tc)
#pragma unroll
    for (int e = 0; e < 8; ++e)
      aop[e * 1024 + tc * 16] = (bf16_t)oc[tc][e];
}

// ---------------------------------------------------------------------------
extern "C" void kernel_launch(void* const* d_in, const int* in_sizes, int n_in,
                              void* d_out, int out_size, void* d_ws, size_t ws_size,
                              hipStream_t stream) {
  const float* x     = (const float*)d_in[0];
  const float* freqs = (const float*)d_in[1];
  const float* Wq    = (const float*)d_in[2];
  const float* Wk    = (const float*)d_in[3];
  const float* Wv    = (const float*)d_in[4];
  const float* Wo    = (const float*)d_in[5];
  float* out = (float*)d_out;

  char* ws = (char*)d_ws;
  const size_t MB = 1024 * 1024;
  bf16_t* xb  = (bf16_t*)(ws + 0);        // 4 MB
  bf16_t* wqt = (bf16_t*)(ws + 4 * MB);   // 2 MB each, transposed (N,K)
  bf16_t* wkt = (bf16_t*)(ws + 6 * MB);
  bf16_t* wvt = (bf16_t*)(ws + 8 * MB);
  bf16_t* wot = (bf16_t*)(ws + 10 * MB);
  float*  Qf  = (float*)(ws + 12 * MB);   // 8 MB each
  float*  Kf  = (float*)(ws + 20 * MB);
  float*  Vf  = (float*)(ws + 28 * MB);
  bf16_t* Qb  = (bf16_t*)(ws + 36 * MB);  // 4 MB each
  bf16_t* Kb  = (bf16_t*)(ws + 40 * MB);
  bf16_t* Vt  = (bf16_t*)(ws + 44 * MB);  // ~4.4 MB
  bf16_t* AOb = (bf16_t*)(ws + 49 * MB);  // 4 MB

  // 1) conversions (weights also transposed -> contiguous K for async staging)
  cvt_f32_bf16<<<8192, 256, 0, stream>>>(x, xb, 2048 * 1024);
  cvt_w_t<<<1024, 256, 0, stream>>>(Wq, wqt);
  cvt_w_t<<<1024, 256, 0, stream>>>(Wk, wkt);
  cvt_w_t<<<1024, 256, 0, stream>>>(Wv, wvt);
  cvt_w_t<<<1024, 256, 0, stream>>>(Wo, wot);

  // 2) Q/K/V projections (WMMA GEMM with async-LDS double buffering)
  dim3 gg(1024 / 128, 2048 / 128);
  gemm_bf16<<<gg, 256, 0, stream>>>(xb, wqt, Qf);
  gemm_bf16<<<gg, 256, 0, stream>>>(xb, wkt, Kf);
  gemm_bf16<<<gg, 256, 0, stream>>>(xb, wvt, Vf);

  // 3) RoPE + head relayout, V transposed+padded
  rope_pack<<<4096, 256, 0, stream>>>(Qf, freqs, Qb);
  rope_pack<<<4096, 256, 0, stream>>>(Kf, freqs, Kb);
  v_pack<<<(2 * 16 * 64 * LP + 255) / 256, 256, 0, stream>>>(Vf, Vt);

  // 4) fused sliding-window attention (WMMA)
  local_attn<<<256, 256, 0, stream>>>(Qb, Kb, Vt, AOb);

  // 5) output projection straight into d_out (f32)
  gemm_bf16<<<gg, 256, 0, stream>>>(AOb, wot, out);
}
